// GTATLayer_25744033972842
// MI455X (gfx1250) — matrix operand order repaired and checked by reading, not compiled, not run on hardware
//
#include <hip/hip_runtime.h>

typedef __attribute__((ext_vector_type(16))) _Float16 v16h;
typedef __attribute__((ext_vector_type(8)))  float    v8f;

#define NB   8
#define NN   2048
#define FDIM 128
#define TDIM 64
#define HID  64
#define NEGV  (-10000.0f)
#define SLOPE 0.01f

__device__ __forceinline__ float lrelu(float x) { return x > 0.0f ? x : SLOPE * x; }
__device__ __forceinline__ int   ic4(const int4& v, int j) {
    return j == 0 ? v.x : j == 1 ? v.y : j == 2 ? v.z : v.w;
}
__device__ __forceinline__ float fc4(const float4& v, int j) {
    return j == 0 ? v.x : j == 1 ? v.y : j == 2 ? v.z : v.w;
}

// ---------------------------------------------------------------------------
// k_prep: fold projection matrices into rank-1 score vectors.
//   wt1 = Wt@t1, wt2 = Wt@t2 (64), wf1 = Wf@f1 (128), plus bias dots.
// consts (floats): [0:64)=wt1 [64:128)=wt2 [128:256)=wf1 [256]=c1 [257]=c2 [258]=cf
// ---------------------------------------------------------------------------
__global__ void k_prep(const float* __restrict__ Wf, const float* __restrict__ bf,
                       const float* __restrict__ Wt, const float* __restrict__ bt,
                       const float* __restrict__ a_feat, const float* __restrict__ a_topo,
                       float* __restrict__ consts) {
    int tid = threadIdx.x;  // 128 threads
    if (tid < 64) {
        float v1 = 0.f, v2 = 0.f;
        for (int h = 0; h < 64; ++h) {
            float w = Wt[tid * 64 + h];
            v1 += w * a_topo[h];
            v2 += w * a_topo[64 + h];
        }
        consts[tid]      = v1;
        consts[64 + tid] = v2;
    }
    {
        float v = 0.f;
        for (int h = 0; h < 64; ++h) v += Wf[tid * 64 + h] * a_feat[h];
        consts[128 + tid] = v;
    }
    if (tid == 0) {
        float c1 = 0.f, c2 = 0.f, cf = 0.f;
        for (int h = 0; h < 64; ++h) {
            c1 += bt[h] * a_topo[h];
            c2 += bt[h] * a_topo[64 + h];
            cf += bf[h] * a_feat[h];
        }
        consts[256] = c1; consts[257] = c2; consts[258] = cf;
    }
}

// ---------------------------------------------------------------------------
// k_proj: one wave per (b,n) row.
//   T_projT[b][h][n] (f16 transposed, WMMA B operand), s1[r], s2[r], hf[r].
// ---------------------------------------------------------------------------
__global__ __launch_bounds__(128) void k_proj(const float* __restrict__ H,
                                              const float* __restrict__ T,
                                              const float* __restrict__ Wt,
                                              const float* __restrict__ bt,
                                              const float* __restrict__ consts,
                                              _Float16* __restrict__ TprojT,
                                              float* __restrict__ s1o,
                                              float* __restrict__ s2o,
                                              float* __restrict__ hfo) {
    int wave = threadIdx.x >> 5, lane = threadIdx.x & 31;
    int r = blockIdx.x * 4 + wave;        // 0..NB*NN-1
    int b = r >> 11, n = r & (NN - 1);
    const float* Tr = T + (size_t)r * TDIM;
    float tlo = Tr[lane], thi = Tr[lane + 32];
    float acc0 = 0.f, acc1 = 0.f, s1a = 0.f, s2a = 0.f;
#pragma unroll
    for (int k = 0; k < 64; ++k) {
        float tk = (k < 32) ? __shfl(tlo, k, 32) : __shfl(thi, k - 32, 32);
        acc0 += tk * Wt[k * 64 + lane];
        acc1 += tk * Wt[k * 64 + lane + 32];
        s1a  += tk * consts[k];
        s2a  += tk * consts[64 + k];
    }
    acc0 += bt[lane];
    acc1 += bt[lane + 32];
    TprojT[((size_t)b * HID + lane)      * NN + n] = (_Float16)acc0;
    TprojT[((size_t)b * HID + lane + 32) * NN + n] = (_Float16)acc1;

    // hf = H row (128) dot wf1 (all lanes compute the full sum redundantly)
    const float* Hr = H + (size_t)r * FDIM;
    float hfa = 0.f;
#pragma unroll
    for (int j = 0; j < 4; ++j) {
        float hv = Hr[lane + 32 * j];
#pragma unroll
        for (int kk = 0; kk < 32; ++kk)
            hfa += __shfl(hv, kk, 32) * consts[128 + 32 * j + kk];
    }
    if (lane == 0) {
        s1o[r] = s1a + consts[256];
        s2o[r] = s2a + consts[257];
        hfo[r] = hfa + consts[258];
    }
}

// ---------------------------------------------------------------------------
// k_stats: per-row online softmax max & sum over 2048 columns.
// One wave per row; each lane consumes 4-wide vectors (float4/int4 B128 loads).
// ---------------------------------------------------------------------------
template <bool MASKED>
__global__ __launch_bounds__(128) void k_stats(const float* __restrict__ sn,
                                               const float* __restrict__ sm,
                                               const int* __restrict__ adj,
                                               const float* __restrict__ biasPtr,
                                               float* __restrict__ rmaxo,
                                               float* __restrict__ rinvo) {
    int wave = threadIdx.x >> 5, lane = threadIdx.x & 31;
    int r = blockIdx.x * 4 + wave;
    int b = r >> 11, n = r & (NN - 1);
    float bias = biasPtr[0];
    float snv  = sn[r];
    const float* smB = sm + (size_t)b * NN;
    const int* adjn = MASKED ? (adj + (size_t)n * NN) : nullptr;
    float rmax = -__builtin_inff(), ssum = 0.f;
    for (int i = 0; i < NN / 128; ++i) {           // 16 iters, 4 elems/lane each
        int m = i * 128 + lane * 4;
        float4 sv = *(const float4*)(smB + m);
        int4 av;
        if (MASKED) av = *(const int4*)(adjn + m);
#pragma unroll
        for (int j = 0; j < 4; ++j) {
            float x = lrelu(snv + fc4(sv, j) + bias);
            if (MASKED) x = ic4(av, j) ? x : NEGV;
            float nm = fmaxf(rmax, x);
            ssum = ssum * __expf(rmax - nm) + __expf(x - nm);
            rmax = nm;
        }
    }
#pragma unroll
    for (int off = 16; off > 0; off >>= 1) {
        float om = __shfl_xor(rmax, off, 32);
        float os = __shfl_xor(ssum, off, 32);
        float nm = fmaxf(rmax, om);
        ssum = ssum * __expf(rmax - nm) + os * __expf(om - nm);
        rmax = nm;
    }
    if (lane == 0) { rmaxo[r] = rmax; rinvo[r] = 1.0f / ssum; }
}

// ---------------------------------------------------------------------------
// k_attn: out[b, 16-row tile, 0:64] = softmax(rank1 scores) @ V  via WMMA f16.
// 8 waves/block, each owns a 256-wide m-stripe; K-chunks of 32; P regenerated
// in registers directly in the ISA A-operand layout; V read f16-transposed
// [b][h][m] so each B operand is two aligned contiguous 16B loads per lane.
// adj is fetched as 4x int4 (B128) per chunk matching the A K-pattern runs.
// ---------------------------------------------------------------------------
template <bool MASKED>
__global__ __launch_bounds__(256) void k_attn(const float* __restrict__ sn,
                                              const float* __restrict__ sm,
                                              const float* __restrict__ rmaxv,
                                              const float* __restrict__ rinvv,
                                              const int* __restrict__ adj,
                                              const _Float16* __restrict__ VT,
                                              const float* __restrict__ biasPtr,
                                              float* __restrict__ out,
                                              _Float16* __restrict__ outT) {
    __shared__ float lds[8][16][64];
    int tid = threadIdx.x;
    int wave = tid >> 5, lane = tid & 31;
    int b = blockIdx.x >> 7;          // NN/16 == 128 tiles
    int ntile = blockIdx.x & 127;
    int nloc = lane & 15;
    int halfw = lane >> 4;
    int n = ntile * 16 + nloc;
    size_t row = (size_t)b * NN + n;
    float snv  = sn[row];
    float rmax = rmaxv[row];
    float bias = biasPtr[0];
    const float* smB = sm + (size_t)b * NN;
    const int* adjn = MASKED ? (adj + (size_t)n * NN) : nullptr;
    int mstart = wave * 256;

    v8f acc[4];
#pragma unroll
    for (int t = 0; t < 4; ++t)
#pragma unroll
        for (int e = 0; e < 8; ++e) acc[t][e] = 0.f;

    for (int c = 0; c < 8; ++c) {
        int mbase = mstart + c * 32;
        float myS = smB[mbase + lane];          // 32 coalesced score loads
        int koff = halfw * 8;
        int4 a0, a1, a2, a3;
        if (MASKED) {                            // two contiguous 8-dword runs
            const int4* ap = (const int4*)(adjn + mbase + koff);
            a0 = ap[0]; a1 = ap[1];              // K = koff+0..7
            a2 = ap[4]; a3 = ap[5];              // K = koff+16..23
        }
        v16h a;
#pragma unroll
        for (int i = 0; i < 16; ++i) {           // A: K=2(v&3)+p+16*(v>=4)+8*half
            int v = i >> 1, p = i & 1;
            int kk = 2 * (v & 3) + p;            // 0..7 within the 8-run
            int k = kk + ((v & 4) ? 16 : 0) + koff;
            float x = lrelu(snv + __shfl(myS, k, 32) + bias);
            if (MASKED) {
                int am = (v & 4) ? ((kk < 4) ? ic4(a2, kk) : ic4(a3, kk - 4))
                                 : ((kk < 4) ? ic4(a0, kk) : ic4(a1, kk - 4));
                x = am ? x : NEGV;
            }
            a[i] = (_Float16)__expf(x - rmax);
        }
#pragma unroll
        for (int t = 0; t < 4; ++t) {            // B: K = 8*half + j (j<8), 16 + 8*half + (j-8) (j>=8)
            const _Float16* bp = VT + ((size_t)b * HID + t * 16 + nloc) * NN
                                              + mbase + halfw * 8;
            v16h bv;
#pragma unroll
            for (int j = 0; j < 8; ++j) { bv[j] = bp[j]; bv[8 + j] = bp[16 + j]; }
            acc[t] = __builtin_amdgcn_wmma_f32_16x16x32_f16(
                false, a, false, bv, (short)0, acc[t], false, false);
        }
    }

    // cross-wave partial reduction through LDS, scale by 1/rowsum, write out
#pragma unroll
    for (int t = 0; t < 4; ++t)
#pragma unroll
        for (int v = 0; v < 8; ++v)
            lds[wave][v + 8 * halfw][t * 16 + nloc] = acc[t][v];
    __syncthreads();
    for (int idx = tid; idx < 1024; idx += 256) {
        int nl = idx >> 6, h = idx & 63;
        size_t orow = (size_t)b * NN + ntile * 16 + nl;
        float s = ((lds[0][nl][h] + lds[1][nl][h]) + (lds[2][nl][h] + lds[3][nl][h]))
                + ((lds[4][nl][h] + lds[5][nl][h]) + (lds[6][nl][h] + lds[7][nl][h]));
        s *= rinvv[orow];
        out[orow * HID + h] = s;
        if (outT) outT[((size_t)b * HID + h) * NN + ntile * 16 + nl] = (_Float16)s;
    }
}

// ---------------------------------------------------------------------------
// k_tf: tf[r] = T_out[r,:] . f2   (one wave per row)
// ---------------------------------------------------------------------------
__global__ __launch_bounds__(128) void k_tf(const float* __restrict__ Tout,
                                            const float* __restrict__ a_feat,
                                            float* __restrict__ tfo) {
    int wave = threadIdx.x >> 5, lane = threadIdx.x & 31;
    int r = blockIdx.x * 4 + wave;
    const float* tr = Tout + (size_t)r * HID;
    float s = tr[lane] * a_feat[64 + lane] + tr[lane + 32] * a_feat[96 + lane];
#pragma unroll
    for (int off = 16; off > 0; off >>= 1) s += __shfl_xor(s, off, 32);
    if (lane == 0) tfo[r] = s;
}

// ---------------------------------------------------------------------------
extern "C" void kernel_launch(void* const* d_in, const int* in_sizes, int n_in,
                              void* d_out, int out_size, void* d_ws, size_t ws_size,
                              hipStream_t stream) {
    const float* H      = (const float*)d_in[0];
    const float* T      = (const float*)d_in[1];
    const int*   adj    = (const int*)  d_in[2];
    const float* Wf     = (const float*)d_in[3];
    const float* bf     = (const float*)d_in[4];
    const float* Wt     = (const float*)d_in[5];
    const float* bt     = (const float*)d_in[6];
    const float* a_feat = (const float*)d_in[7];
    const float* b_feat = (const float*)d_in[8];
    const float* a_topo = (const float*)d_in[9];
    const float* b_topo = (const float*)d_in[10];

    float* Hout = (float*)d_out;
    float* Tout = Hout + (size_t)NB * NN * HID;

    char* W = (char*)d_ws;
    float*    consts = (float*)W;                                   // 4 KB
    _Float16* TprojT = (_Float16*)(W + 4096);                       // 2 MB
    float* s1     = (float*)(W + 4096 + 2u * NB * HID * NN);
    float* s2     = s1 + NB * NN;
    float* hf     = s2 + NB * NN;
    float* tf     = hf + NB * NN;
    float* rmax_t = tf + NB * NN;
    float* rinv_t = rmax_t + NB * NN;
    float* rmax_f = rinv_t + NB * NN;
    float* rinv_f = rmax_f + NB * NN;
    _Float16* ToutT = (_Float16*)(rinv_f + NB * NN);                // 2 MB

    k_prep<<<1, 128, 0, stream>>>(Wf, bf, Wt, bt, a_feat, a_topo, consts);
    k_proj<<<NB * NN / 4, 128, 0, stream>>>(H, T, Wt, bt, consts, TprojT, s1, s2, hf);
    k_stats<true><<<NB * NN / 4, 128, 0, stream>>>(s1, s2, adj, b_topo, rmax_t, rinv_t);
    k_attn<true><<<NB * (NN / 16), 256, 0, stream>>>(s1, s2, rmax_t, rinv_t, adj,
                                                     TprojT, b_topo, Tout, ToutT);
    k_tf<<<NB * NN / 4, 128, 0, stream>>>(Tout, a_feat, tf);
    k_stats<false><<<NB * NN / 4, 128, 0, stream>>>(hf, tf, nullptr, b_feat, rmax_f, rinv_f);
    k_attn<false><<<NB * (NN / 16), 256, 0, stream>>>(hf, tf, rmax_f, rinv_f, nullptr,
                                                      ToutT, b_feat, Hout, nullptr);
}
